// SpikingEncoderBlock_4999341933019
// MI455X (gfx1250) — compile-verified, matrix operand order checked by reading, not run on hardware
//
#include <hip/hip_runtime.h>

typedef __attribute__((ext_vector_type(16))) _Float16 v16h;
typedef __attribute__((ext_vector_type(8)))  float    v8f;

#define C_IN    32
#define C_OUT   64
#define IN_D    64
#define OUT_D   32
#define SP_OUT  (OUT_D*OUT_D*OUT_D)   // 32768 spatial sites per batch
#define NSITES  (2*SP_OUT)            // 65536 total sites
#define NTAPS   27                    // 3x3x3 taps; K = NTAPS*C_IN = 864
#define TSTEPS  10
#define KTOP    6                     // max(1, int(0.1*64))
#define TAU_U   0.9f
#define V_TH    1.0f
#define BN_EPS  1e-5f

#define ASLICE_BYTES 4096             // one K-chunk of packed A: 4*32*16 f16 = 2048 halves

// d_ws layout (bytes): packed A fragments then BN scale/shift
#define WS_A_OFF      0        // 27*4096 B = 110592 B (32B-aligned frags)
#define WS_SCALE_OFF  110592   // 64 floats
#define WS_SHIFT_OFF  110848   // 64 floats

// ---------------------------------------------------------------------------
// Pack fp32 weights [64][32][3][3][3] into the CDNA5 16-bit A-fragment layout:
// frag index = ((chunk*4 + mt)*32 + lane)*16 + e, chunk = tap (kd*9+kh*3+kw).
// lanes 0-15: row m=mt*16+lane,    K: e<8 -> e,   e>=8 -> e+8   (ci 0..7,16..23)
// lanes16-31: row m=mt*16+lane-16, K: e<8 -> e+8, e>=8 -> e+16  (ci 8..15,24..31)
// ---------------------------------------------------------------------------
__global__ void pack_weights_kernel(const float* __restrict__ w,
                                    _Float16* __restrict__ pA) {
  int id = blockIdx.x * 256 + threadIdx.x;
  if (id >= NTAPS * C_OUT * C_IN) return;          // 55296
  int e     = id & 15;
  int lane  = (id >> 4) & 31;
  int mt    = (id >> 9) & 3;
  int chunk = id >> 11;
  int m  = mt * 16 + (lane & 15);
  int ci = (lane < 16) ? ((e < 8) ? e : e + 8)
                       : ((e < 8) ? e + 8 : e + 16);
  pA[id] = (_Float16)w[(m * C_IN + ci) * NTAPS + chunk];
}

// ---------------------------------------------------------------------------
// Implicit-GEMM conv3d via v_wmma_f32_16x16x32_f16.
// Block = 128 threads (4 waves). Wave mt owns a 16-channel M-tile; all waves
// share one 16-site N-tile. K loop: 27 chunks of 32 (one tap x 32 cin).
// Packed-A K-chunks are double-buffered in LDS via async global->LDS copies
// (ASYNCcnt-tracked, s_wait_asynccnt); B fragments are VALU-gathered
// (f32->f16 + zero pad) into LDS and read back as v16h (ds_load_b128 x2).
// ---------------------------------------------------------------------------
__global__ void conv_wmma_kernel(const float* __restrict__ x,
                                 const _Float16* __restrict__ pA,
                                 const float* __restrict__ bias,
                                 float* __restrict__ y) {
  __shared__ __align__(32) _Float16 ldsA[2 * (ASLICE_BYTES / 2)];  // 2 x 4KB
  __shared__ __align__(32) _Float16 ldsB[512];                     // 32 lanes x 16 halves
  const int tid      = threadIdx.x;
  const int lane     = tid & 31;
  const int mt       = tid >> 5;                   // 0..3 -> 16-channel M tile
  const int siteBase = blockIdx.x * 16;

  const unsigned ldsABase = (unsigned)(uintptr_t)(void*)ldsA;

  // prologue: async-copy chunk 0's packed-A slice into buffer 0
  {
#pragma unroll
    for (int r = 0; r < 2; ++r) {
      unsigned goff = (unsigned)(r * 2048 + tid * 16);             // bytes into pA
      unsigned loff = ldsABase + (unsigned)(r * 2048 + tid * 16);  // LDS byte addr
      asm volatile("global_load_async_to_lds_b128 %0, %1, %2"
                   :: "v"(loff), "v"(goff), "s"(pA) : "memory");
    }
  }

  v8f acc = {};

  for (int chunk = 0; chunk < NTAPS; ++chunk) {
    const int kd = chunk / 9, kh = (chunk / 3) % 3, kw = chunk % 3;

    // cooperative B-fragment fill: 128 threads x 4 entries = 512 halves
#pragma unroll
    for (int j = 0; j < 4; ++j) {
      int idx = tid * 4 + j;
      int bl  = idx >> 4;                          // fragment lane 0..31
      int e   = idx & 15;
      int n   = bl & 15;                           // column (site in tile)
      int ci  = (bl < 16) ? ((e < 8) ? e : e + 8)
                          : ((e < 8) ? e + 8 : e + 16);
      int s   = siteBase + n;
      int b   = s >> 15;
      int sp  = s & (SP_OUT - 1);
      int od  = sp >> 10, oh = (sp >> 5) & 31, ow = sp & 31;
      int idd = 2 * od + kd - 1;
      int ih  = 2 * oh + kh - 1;
      int iw  = 2 * ow + kw - 1;
      float v = 0.0f;
      if ((unsigned)idd < 64u && (unsigned)ih < 64u && (unsigned)iw < 64u)
        v = x[(((b * C_IN + ci) * IN_D + idd) * IN_D + ih) * IN_D + iw];
      ldsB[idx] = (_Float16)v;
    }

    // issue async copy of the NEXT chunk's A slice into the other buffer
    if (chunk + 1 < NTAPS) {
      const unsigned nbuf = (unsigned)((chunk + 1) & 1) * ASLICE_BYTES;
#pragma unroll
      for (int r = 0; r < 2; ++r) {
        unsigned goff = (unsigned)((chunk + 1) * ASLICE_BYTES + r * 2048 + tid * 16);
        unsigned loff = ldsABase + nbuf + (unsigned)(r * 2048 + tid * 16);
        asm volatile("global_load_async_to_lds_b128 %0, %1, %2"
                     :: "v"(loff), "v"(goff), "s"(pA) : "memory");
      }
      // drain the previous slice's 2 copies; keep the 2 just issued in flight
      asm volatile("s_wait_asynccnt 0x2" ::: "memory");
    } else {
      asm volatile("s_wait_asynccnt 0x0" ::: "memory");
    }
    __syncthreads();   // A slice + B fragment visible to all waves

    const _Float16* aSlice = ldsA + (size_t)(chunk & 1) * (ASLICE_BYTES / 2);
    v16h bFrag = *(const v16h*)(ldsB + lane * 16);                 // ds_load_b128 x2
    v16h aFrag = *(const v16h*)(aSlice + ((mt * 32 + lane) << 4)); // ds_load_b128 x2
    acc = __builtin_amdgcn_wmma_f32_16x16x32_f16(
        /*neg_a=*/false, aFrag, /*neg_b=*/false, bFrag,
        /*c_mod=*/(short)0, acc, /*reuse_a=*/false, /*reuse_b=*/false);
    __syncthreads();   // before next iteration overwrites ldsB / ldsA buffer
  }

  // epilogue: C layout — VGPR j: lanes 0-15 hold M=j, lanes 16-31 hold M=j+8
  const int n  = lane & 15;
  const int s  = siteBase + n;
  const int b  = s >> 15;
  const int sp = s & (SP_OUT - 1);
#pragma unroll
  for (int j = 0; j < 8; ++j) {
    int c = mt * 16 + ((lane < 16) ? j : j + 8);
    y[(size_t)(b * C_OUT + c) * SP_OUT + sp] = acc[j] + bias[c];
  }
}

// ---------------------------------------------------------------------------
// Per-channel batch stats -> fused BN scale/shift. One block per channel.
// ---------------------------------------------------------------------------
__global__ void bn_stats_kernel(const float* __restrict__ y,
                                const float* __restrict__ gamma,
                                const float* __restrict__ beta,
                                float* __restrict__ scale,
                                float* __restrict__ shift) {
  __shared__ float sSum[256];
  __shared__ float sSq[256];
  const int c = blockIdx.x;
  const int tid = threadIdx.x;
  float sum = 0.f, sq = 0.f;
  for (int i = tid; i < NSITES; i += 256) {
    int b  = i >> 15;
    int sp = i & (SP_OUT - 1);
    float v = y[(size_t)(b * C_OUT + c) * SP_OUT + sp];
    sum += v; sq += v * v;
  }
  sSum[tid] = sum; sSq[tid] = sq;
  __syncthreads();
  for (int off = 128; off > 0; off >>= 1) {
    if (tid < off) { sSum[tid] += sSum[tid + off]; sSq[tid] += sSq[tid + off]; }
    __syncthreads();
  }
  if (tid == 0) {
    float mean = sSum[0] / (float)NSITES;
    float var  = sSq[0] / (float)NSITES - mean * mean;
    float sc   = gamma[c] * rsqrtf(var + BN_EPS);
    scale[c] = sc;
    shift[c] = beta[c] - mean * sc;
  }
}

// ---------------------------------------------------------------------------
// Fused 10-step LIF loop. Thread per site; 64-channel membrane/input/rate
// state in registers; top-6 threshold via unrolled insertion network.
// `rates` aliases `y`: each y element is read only by the one thread that
// later overwrites it with its rate, so there is no cross-thread hazard.
// ---------------------------------------------------------------------------
__global__ void spike_kernel(const float* __restrict__ y,
                             const float* __restrict__ scale,
                             const float* __restrict__ shift,
                             float* __restrict__ spikes,   // [B,T,C,sp]
                             float* __restrict__ rates) {  // [B,C,sp]
  __shared__ float sSc[C_OUT];
  __shared__ float sSh[C_OUT];
  const int tid = threadIdx.x;
  if (tid < C_OUT) { sSc[tid] = scale[tid]; sSh[tid] = shift[tid]; }
  __syncthreads();

  const int s  = blockIdx.x * 256 + tid;
  const int b  = s >> 15;
  const int sp = s & (SP_OUT - 1);

  float yn[C_OUT], u[C_OUT], rate[C_OUT];
#pragma unroll
  for (int c = 0; c < C_OUT; ++c) {
    yn[c]   = y[(size_t)(b * C_OUT + c) * SP_OUT + sp] * sSc[c] + sSh[c];
    u[c]    = 0.f;
    rate[c] = 0.f;
  }

#pragma unroll 1
  for (int t = 0; t < TSTEPS; ++t) {
    // kth-largest (k=6) of the 64 membrane values
    float top[KTOP];
#pragma unroll
    for (int j = 0; j < KTOP; ++j) top[j] = -3.0e38f;
#pragma unroll
    for (int c = 0; c < C_OUT; ++c) {
      float v = u[c];
#pragma unroll
      for (int j = 0; j < KTOP; ++j) {
        float hi = fmaxf(top[j], v);
        v        = fminf(top[j], v);
        top[j]   = hi;
      }
    }
    const float kth = top[KTOP - 1];

    float* outT = spikes + (size_t)(b * TSTEPS + t) * (C_OUT * SP_OUT) + sp;
#pragma unroll
    for (int c = 0; c < C_OUT; ++c) {
      float uu = (u[c] >= kth) ? u[c] : 0.f;   // lateral inhibition
      uu = TAU_U * uu + yn[c];                 // leaky integrate
      float sv = (uu >= V_TH) ? 1.f : 0.f;     // fire
      uu -= sv * V_TH;                         // soft reset
      u[c] = uu;
      rate[c] += sv;
      outT[(size_t)c * SP_OUT] = sv;           // coalesced over sp across lanes
    }
  }

#pragma unroll
  for (int c = 0; c < C_OUT; ++c)
    rates[(size_t)(b * C_OUT + c) * SP_OUT + sp] = rate[c] * (1.0f / TSTEPS);
}

extern "C" void kernel_launch(void* const* d_in, const int* in_sizes, int n_in,
                              void* d_out, int out_size, void* d_ws, size_t ws_size,
                              hipStream_t stream) {
  const float* x     = (const float*)d_in[0];
  const float* w     = (const float*)d_in[1];
  const float* bias  = (const float*)d_in[2];
  const float* gamma = (const float*)d_in[3];
  const float* beta  = (const float*)d_in[4];

  float* out    = (float*)d_out;
  float* spikes = out;                                             // 41,943,040 floats
  float* yrates = out + (size_t)2 * TSTEPS * C_OUT * SP_OUT;       // rates region, reused as y

  _Float16* pA   = (_Float16*)((char*)d_ws + WS_A_OFF);
  float*    scl  = (float*)((char*)d_ws + WS_SCALE_OFF);
  float*    shf  = (float*)((char*)d_ws + WS_SHIFT_OFF);

  pack_weights_kernel<<<(NTAPS * C_OUT * C_IN + 255) / 256, 256, 0, stream>>>(w, pA);
  conv_wmma_kernel<<<NSITES / 16, 128, 0, stream>>>(x, pA, bias, yrates);
  bn_stats_kernel<<<C_OUT, 256, 0, stream>>>(yrates, gamma, beta, scl, shf);
  spike_kernel<<<NSITES / 256, 256, 0, stream>>>(yrates, scl, shf, spikes, yrates);
}
